// Decoder_42382737277570
// MI455X (gfx1250) — matrix operand order repaired and checked by reading, not compiled, not run on hardware
//
#include <hip/hip_runtime.h>
#include <hip/hip_bf16.h>

typedef __attribute__((ext_vector_type(16))) _Float16 v16h;
typedef __attribute__((ext_vector_type(8)))  float    v8f;
typedef _Float16 half_t;

#define H_      32
#define DH_     128
#define D_      4096
#define S_      16384
#define DFF_    11008
#define NKEYS   16385            // S+1
#define NREMAIN 328              // max(128, 16385 - int(16385*0.98))

// ---- d_out layout (4-byte element offsets) ----
#define OUT_OFF   ((size_t)0)                               // 4096 f32
#define KEYS_OFF  ((size_t)4096)                            // 32*16385*128 f32
#define VALS_OFF  (KEYS_OFF + (size_t)H_*NKEYS*DH_)         // same size
#define HASH_OFF  (VALS_OFF + (size_t)H_*NKEYS*DH_)         // 32*16385*16 i32

// ---- workspace layout (float-slot offsets) ----
#define OFF_XH    ((size_t)0)        // 4096 halves  (2048 slots)
#define OFF_QKV   ((size_t)4096)     // 12288 f32: q@+0, k@+4096, v@+8192
#define OFF_QHASH ((size_t)16384)    // 512 i32
#define OFF_SIM   ((size_t)17408)    // 32*16385 i32 = 524320
#define OFF_THR   ((size_t)544000)   // 128 i32
#define OFF_IDX   ((size_t)544256)   // 32*328 i32 = 10496
#define OFF_O     ((size_t)556000)   // 4096 f32
#define OFF_OH    ((size_t)560096)   // 4096 halves (2048 slots)
#define OFF_HBUF  ((size_t)562144)   // 4096 f32
#define OFF_YH    ((size_t)566240)   // 4096 halves (2048 slots)
#define OFF_G     ((size_t)568288)   // 11008 f32
#define OFF_U     ((size_t)579296)   // 11008 f32
#define OFF_ACTH  ((size_t)590304)   // 11008 halves (5504 slots)

// ---------------------------------------------------------------------------
// RMSNorm: single block, 256 threads.
// ---------------------------------------------------------------------------
__global__ void rmsnorm_k(const float* __restrict__ in, const float* __restrict__ w,
                          float* __restrict__ outf, half_t* __restrict__ outh, int n) {
  __shared__ float red[256];
  int t = threadIdx.x;
  float s = 0.f;
  for (int i = t; i < n; i += 256) { float v = in[i]; s += v * v; }
  red[t] = s; __syncthreads();
  for (int o = 128; o > 0; o >>= 1) { if (t < o) red[t] += red[t + o]; __syncthreads(); }
  float r = rsqrtf(red[0] / (float)n + 1e-5f);
  for (int i = t; i < n; i += 256) {
    float v = w[i] * in[i] * r;
    if (outf) outf[i] = v;
    outh[i] = (half_t)v;
  }
}

// ---------------------------------------------------------------------------
// GEMV via V_WMMA_F32_16X16X32_F16, K unrolled x2 with independent
// accumulators (avoids the WMMA->WMMA C-operand RAW hazard chain).
// out[j] = sum_i x[i]*W[i*N+j] (+resid). One wave -> 16 output columns.
// ---------------------------------------------------------------------------
__device__ __forceinline__ v16h build_a(const half_t* __restrict__ xv, int k0,
                                        int lane, int lm) {
  v16h a = {};
  if (lm == 0) {
#pragma unroll
    for (int h = 0; h < 16; ++h) {
      int kk = ((lane < 16) ? 0 : 16) + h;
      a[h] = xv[k0 + kk];
    }
  }
  return a;
}

__device__ __forceinline__ v16h build_b(const float* __restrict__ wp, size_t N) {
  v16h b;
#pragma unroll
  for (int h = 0; h < 16; ++h) b[h] = (half_t)wp[(size_t)h * N];
  return b;
}

__global__ void gemv_wmma_k(const half_t* __restrict__ xv, const float* __restrict__ W,
                            int K, int N, float* __restrict__ out,
                            const float* __restrict__ resid) {
  int wave = (int)((blockIdx.x * blockDim.x + threadIdx.x) >> 5);
  int lane = threadIdx.x & 31;
  int col0 = wave * 16;
  if (col0 >= N) return;
  int lm  = lane & 15;
  int col = col0 + lm;
  int kb  = (lane < 16) ? 0 : 16;
  v8f c0 = {}, c1 = {};
  for (int k0 = 0; k0 < K; k0 += 64) {                 // K % 64 == 0 for all calls
    const float* wp0 = W + (size_t)(k0 + kb) * N + col;
    const float* wp1 = wp0 + (size_t)32 * N;
    v16h a0 = build_a(xv, k0,      lane, lm);
    v16h b0 = build_b(wp0, (size_t)N);
    v16h a1 = build_a(xv, k0 + 32, lane, lm);
    v16h b1 = build_b(wp1, (size_t)N);
    if (k0 + 64 < K) __builtin_prefetch(wp0 + (size_t)64 * N, 0, 1);  // global_prefetch_b8
    c0 = __builtin_amdgcn_wmma_f32_16x16x32_f16(false, a0, false, b0, (short)0, c0,
                                                false, false);
    c1 = __builtin_amdgcn_wmma_f32_16x16x32_f16(false, a1, false, b1, (short)0, c1,
                                                false, false);
  }
  if (lane < 16) {                      // C layout: VGPR0, lanes 0-15 = M=0,N=lane
    float v = c0[0] + c1[0];
    if (resid) v += resid[col0 + lane];
    out[col0 + lane] = v;
  }
}

// ---------------------------------------------------------------------------
// Per-head RoPE (pos = 16384) + hash_code for q and roped k; appends roped k,
// v and k_hash row NKEYS-1; writes q_hash; persists roped q for attention.
// 32 blocks x 128 threads.
// ---------------------------------------------------------------------------
__global__ void rope_hash_k(float* __restrict__ qkv, const float* __restrict__ rot1,
                            const float* __restrict__ rot2,
                            float* __restrict__ keys_out, float* __restrict__ vals_out,
                            int* __restrict__ hash_out, int* __restrict__ qhash_out) {
  int h = blockIdx.x, d = threadIdx.x;
  __shared__ float qs[128], ks[128], ybuf[128];
  __shared__ int bits[128];
  float* qp = qkv + (size_t)h * 128;
  float* kp = qkv + 4096 + (size_t)h * 128;
  const float* vp = qkv + 8192 + (size_t)h * 128;

  float ang = 16384.0f * powf(10000.0f, -(float)((d & 63) * 2) / 128.0f);
  float sn, cs; sincosf(ang, &sn, &cs);
  float qr = (d < 64) ? -qp[d + 64] : qp[d - 64];
  float kr = (d < 64) ? -kp[d + 64] : kp[d - 64];
  qs[d] = qp[d] * cs + qr * sn;
  ks[d] = kp[d] * cs + kr * sn;
  __syncthreads();

  qp[d] = qs[d];                                            // roped q for attention
  size_t slot = ((size_t)h * NKEYS + (NKEYS - 1)) * 128 + d;
  keys_out[slot] = ks[d];
  vals_out[slot] = vp[d];

  const float* r1 = rot1 + (size_t)h * 128 * 128 + d;
  const float* r2 = rot2 + (size_t)h * 128 * 128 + d;

  // ---- q hash ----
  float acc = 0.f;
  for (int i = 0; i < 128; ++i) acc += qs[i] * r1[(size_t)i * 128];
  ybuf[d] = acc / (1.f + expf(-acc));                       // silu
  __syncthreads();
  acc = 0.f;
  for (int i = 0; i < 128; ++i) acc += ybuf[i] * r2[(size_t)i * 128];
  bits[d] = (acc > 0.f) ? 1 : 0;
  __syncthreads();
  if (d < 16) {
    int byte = 0;
    for (int j = 0; j < 8; ++j) byte |= bits[d * 8 + j] << j;
    qhash_out[h * 16 + d] = byte;
  }
  __syncthreads();

  // ---- k hash ----
  acc = 0.f;
  for (int i = 0; i < 128; ++i) acc += ks[i] * r1[(size_t)i * 128];
  ybuf[d] = acc / (1.f + expf(-acc));
  __syncthreads();
  acc = 0.f;
  for (int i = 0; i < 128; ++i) acc += ybuf[i] * r2[(size_t)i * 128];
  bits[d] = (acc > 0.f) ? 1 : 0;
  __syncthreads();
  if (d < 16) {
    int byte = 0;
    for (int j = 0; j < 8; ++j) byte |= bits[d * 8 + j] << j;
    hash_out[((size_t)h * NKEYS + (NKEYS - 1)) * 16 + d] = byte;
  }
}

// ---------------------------------------------------------------------------
// Hamming sims + per-head histogram -> threshold (T, G=#>T, R=328-G).
// ---------------------------------------------------------------------------
__global__ void sim_hist_k(const int* __restrict__ hash_all, const int* __restrict__ qhash,
                           int* __restrict__ sim, int* __restrict__ thr) {
  int h = blockIdx.x, t = threadIdx.x;
  __shared__ int qh[16];
  __shared__ int hist[129];
  if (t < 16)  qh[t] = qhash[h * 16 + t];
  if (t < 129) hist[t] = 0;
  __syncthreads();
  const int* base = hash_all + (size_t)h * NKEYS * 16;
  int* simh = sim + (size_t)h * NKEYS;
  for (int s = t; s < NKEYS; s += 256) {
    const int4* p = (const int4*)(base + (size_t)s * 16);
    int pc = 0;
#pragma unroll
    for (int c = 0; c < 4; ++c) {
      int4 v = p[c];
      pc += __popc((v.x ^ qh[c * 4 + 0]) & 0xff);
      pc += __popc((v.y ^ qh[c * 4 + 1]) & 0xff);
      pc += __popc((v.z ^ qh[c * 4 + 2]) & 0xff);
      pc += __popc((v.w ^ qh[c * 4 + 3]) & 0xff);
    }
    int sv = 128 - pc;
    simh[s] = sv;
    atomicAdd(&hist[sv], 1);
  }
  __syncthreads();
  if (t == 0) {
    int acc = 0, T = 0, G = 0, R = NREMAIN;
    for (int v = 128; v >= 0; --v) {
      if (acc + hist[v] >= NREMAIN) { T = v; G = acc; R = NREMAIN - acc; break; }
      acc += hist[v];
    }
    thr[h * 4 + 0] = T; thr[h * 4 + 1] = G; thr[h * 4 + 2] = R;
  }
}

// ---------------------------------------------------------------------------
// Ordered ballot/prefix selection: all sim>T, then lowest-index sim==T up to R.
// ---------------------------------------------------------------------------
__global__ void select_k(const int* __restrict__ sim, const int* __restrict__ thr,
                         int* __restrict__ idx) {
  int h = blockIdx.x, lane = threadIdx.x;
  int T = thr[h * 4 + 0], G = thr[h * 4 + 1], R = thr[h * 4 + 2];
  const int* simh = sim + (size_t)h * NKEYS;
  int* out = idx + h * NREMAIN;
  int bg = 0, be = 0;
  for (int base = 0; base < NKEYS; base += 32) {
    int s = base + lane;
    bool valid = s < NKEYS;
    int sv = valid ? simh[valid ? s : 0] : -1;
    unsigned mg = (unsigned)__ballot(valid && sv > T);
    unsigned me = (unsigned)__ballot(valid && sv == T);
    unsigned lt = (lane == 0) ? 0u : (0xffffffffu >> (32 - lane));
    if (valid && sv > T) out[bg + __popc(mg & lt)] = s;
    if (valid && sv == T) { int r = be + __popc(me & lt); if (r < R) out[G + r] = s; }
    bg += __popc(mg); be += __popc(me);
  }
}

// ---------------------------------------------------------------------------
// Gathered attention over 328 selected keys. 32 blocks x 128 threads.
// ---------------------------------------------------------------------------
__global__ void attn_k(const float* __restrict__ qrot, const int* __restrict__ idx,
                       const float* __restrict__ keys, const float* __restrict__ vals,
                       float* __restrict__ o, half_t* __restrict__ oh) {
  int h = blockIdx.x, t = threadIdx.x;
  __shared__ float qsh[128];
  __shared__ float sc[NREMAIN];
  __shared__ int   sidx[NREMAIN];
  __shared__ float red[128];
  qsh[t] = qrot[(size_t)h * 128 + t];
  __syncthreads();
  const float scale = 0.08838834764831845f;                 // 1/sqrt(128)
  float lmax = -1e30f;
  for (int kk = t; kk < NREMAIN; kk += 128) {
    int s = idx[h * NREMAIN + kk];
    sidx[kk] = s;
    const float* kr = keys + ((size_t)h * NKEYS + s) * 128;
    float acc = 0.f;
    for (int d4 = 0; d4 < 128; d4 += 4) {
      float4 kv = *(const float4*)(kr + d4);
      acc += qsh[d4] * kv.x + qsh[d4 + 1] * kv.y + qsh[d4 + 2] * kv.z + qsh[d4 + 3] * kv.w;
    }
    acc *= scale;
    sc[kk] = acc;
    lmax = fmaxf(lmax, acc);
  }
  red[t] = lmax; __syncthreads();
  for (int off = 64; off > 0; off >>= 1) { if (t < off) red[t] = fmaxf(red[t], red[t + off]); __syncthreads(); }
  float m = red[0]; __syncthreads();
  float lsum = 0.f;
  for (int kk = t; kk < NREMAIN; kk += 128) { float e = expf(sc[kk] - m); sc[kk] = e; lsum += e; }
  red[t] = lsum; __syncthreads();
  for (int off = 64; off > 0; off >>= 1) { if (t < off) red[t] += red[t + off]; __syncthreads(); }
  float inv = 1.f / red[0];
  __syncthreads();
  float acc = 0.f;
  for (int kk = 0; kk < NREMAIN; ++kk)
    acc += sc[kk] * vals[((size_t)h * NKEYS + sidx[kk]) * 128 + t];
  acc *= inv;
  o [(size_t)h * 128 + t] = acc;
  oh[(size_t)h * 128 + t] = (half_t)acc;
}

// ---------------------------------------------------------------------------
// Bulk cache copy via CDNA5 async Global->LDS->Global path (no VGPR data).
// Each lane moves `perThread` 16-byte chunks through a double-buffered LDS
// slot pair; ASYNCcnt tracks completion. The s_wait_asynccnt 0 before the
// store of slot s also retires the previous slot's store, so load(B) overlaps
// store(A). Chunk c -> head = c>>headShift, rem = c & mask (per-head remap).
// ---------------------------------------------------------------------------
__global__ void copy_async_k(const char* __restrict__ src, char* __restrict__ dst,
                             int headShift, size_t srcPerHeadB, size_t dstPerHeadB,
                             int perThread, size_t chunkStride) {
  __shared__ char buf[2 * 256 * 16];
  int t = threadIdx.x;
  unsigned lds0 = (unsigned)(size_t)&buf[t * 16];
  unsigned lds1 = lds0 + 4096u;
  size_t gtid = (size_t)blockIdx.x * blockDim.x + t;
  size_t mask = ((size_t)1 << headShift) - 1;
  for (int i = 0; i < perThread; ++i) {
    size_t c = gtid + (size_t)i * chunkStride;
    size_t head = c >> headShift;
    size_t rem  = c & mask;
    unsigned long long s = (unsigned long long)(size_t)(src + head * srcPerHeadB + rem * 16);
    unsigned long long d = (unsigned long long)(size_t)(dst + head * dstPerHeadB + rem * 16);
    unsigned l = (i & 1) ? lds1 : lds0;
    asm volatile("global_load_async_to_lds_b128 %0, %1, off"
                 :: "v"(l), "v"(s) : "memory");
    asm volatile("s_wait_asynccnt 0x0" ::: "memory");
    asm volatile("global_store_async_from_lds_b128 %0, %1, off"
                 :: "v"(d), "v"(l) : "memory");
  }
  asm volatile("s_wait_asynccnt 0x0" ::: "memory");
}

__global__ void silumul_k(const float* __restrict__ g, const float* __restrict__ u,
                          half_t* __restrict__ acth, int n) {
  int i = blockIdx.x * blockDim.x + threadIdx.x;
  if (i < n) {
    float gv = g[i];
    acth[i] = (half_t)(gv / (1.f + expf(-gv)) * u[i]);
  }
}

// ---------------------------------------------------------------------------
extern "C" void kernel_launch(void* const* d_in, const int* in_sizes, int n_in,
                              void* d_out, int out_size, void* d_ws, size_t ws_size,
                              hipStream_t stream) {
  (void)in_sizes; (void)n_in; (void)out_size; (void)ws_size;
  const float* hidden = (const float*)d_in[0];
  const float* kcache = (const float*)d_in[1];
  const float* vcache = (const float*)d_in[2];
  const int*   hcache = (const int*)  d_in[3];
  const float* Wq = (const float*)d_in[4];
  const float* Wk = (const float*)d_in[5];
  const float* Wv = (const float*)d_in[6];
  const float* Wo = (const float*)d_in[7];
  const float* rot1 = (const float*)d_in[8];
  const float* rot2 = (const float*)d_in[9];
  const float* ln1  = (const float*)d_in[10];
  const float* ln2  = (const float*)d_in[11];
  const float* Wg = (const float*)d_in[12];
  const float* Wu = (const float*)d_in[13];
  const float* Wd = (const float*)d_in[14];

  float* outv = (float*)d_out;
  float* keys = outv + KEYS_OFF;
  float* vals = outv + VALS_OFF;
  int*   hash = (int*)d_out + HASH_OFF;
  float* ws   = (float*)d_ws;

  half_t* xh   = (half_t*)(ws + OFF_XH);
  float*  qkv  = ws + OFF_QKV;
  int*    qhsh = (int*)(ws + OFF_QHASH);
  int*    sim  = (int*)(ws + OFF_SIM);
  int*    thr  = (int*)(ws + OFF_THR);
  int*    idx  = (int*)(ws + OFF_IDX);
  float*  obuf = ws + OFF_O;
  half_t* oh   = (half_t*)(ws + OFF_OH);
  float*  hbuf = ws + OFF_HBUF;
  half_t* yh   = (half_t*)(ws + OFF_YH);
  float*  gbuf = ws + OFF_G;
  float*  ubuf = ws + OFF_U;
  half_t* acth = (half_t*)(ws + OFF_ACTH);

  // 1) bulk cache copies via async Global<->LDS DMA path
  //    keys/vals: 16,777,216 chunks of 16B each; 8 chunks/thread -> 8192 blocks
  copy_async_k<<<8192, 256, 0, stream>>>((const char*)kcache, (char*)keys,
                                         19, (size_t)524288 * 16, (size_t)524320 * 16,
                                         8, (size_t)8192 * 256);
  copy_async_k<<<8192, 256, 0, stream>>>((const char*)vcache, (char*)vals,
                                         19, (size_t)524288 * 16, (size_t)524320 * 16,
                                         8, (size_t)8192 * 256);
  //    hash: 2,097,152 chunks; 4 chunks/thread -> 2048 blocks
  copy_async_k<<<2048, 256, 0, stream>>>((const char*)hcache, (char*)hash,
                                         16, (size_t)65536 * 16, (size_t)65540 * 16,
                                         4, (size_t)2048 * 256);

  // 2) rmsnorm1 -> x (half)
  rmsnorm_k<<<1, 256, 0, stream>>>(hidden, ln1, nullptr, xh, D_);

  // 3) QKV GEMVs (WMMA). N=4096 -> 256 waves -> 32 blocks of 256 threads.
  gemv_wmma_k<<<32, 256, 0, stream>>>(xh, Wq, D_, D_, qkv,        nullptr);
  gemv_wmma_k<<<32, 256, 0, stream>>>(xh, Wk, D_, D_, qkv + 4096, nullptr);
  gemv_wmma_k<<<32, 256, 0, stream>>>(xh, Wv, D_, D_, qkv + 8192, nullptr);

  // 4) RoPE + hash, append new k/v/hash rows, persist roped q, write q_hash
  rope_hash_k<<<H_, 128, 0, stream>>>(qkv, rot1, rot2, keys, vals, hash, qhsh);

  // 5) Hamming sims + histogram threshold; 6) ordered exact selection
  sim_hist_k<<<H_, 256, 0, stream>>>(hash, qhsh, sim, thr);
  select_k  <<<H_,  32, 0, stream>>>(sim, thr, idx);

  // 7) gathered attention
  attn_k<<<H_, 128, 0, stream>>>(qkv, idx, keys, vals, obuf, oh);

  // 8) output projection + residual -> h
  gemv_wmma_k<<<32, 256, 0, stream>>>(oh, Wo, D_, D_, hbuf, hidden);

  // 9) rmsnorm2 -> y (half)
  rmsnorm_k<<<1, 256, 0, stream>>>(hbuf, ln2, nullptr, yh, D_);

  // 10) MLP: g = y@Wg, u = y@Wu (N=11008 -> 688 waves -> 86 blocks)
  gemv_wmma_k<<<86, 256, 0, stream>>>(yh, Wg, D_, DFF_, gbuf, nullptr);
  gemv_wmma_k<<<86, 256, 0, stream>>>(yh, Wu, D_, DFF_, ubuf, nullptr);
  silumul_k<<<(DFF_ + 255) / 256, 256, 0, stream>>>(gbuf, ubuf, acth, DFF_);

  // 11) down-proj + residual -> final out (K=11008 = 172 unrolled k-steps)
  gemv_wmma_k<<<32, 256, 0, stream>>>(acth, Wd, DFF_, D_, outv, hbuf);
}
